// VA_aware_SupConLoss_45260365365260
// MI455X (gfx1250) — matrix operand order repaired and checked by reading, not compiled, not run on hardware
//
#include <hip/hip_runtime.h>
#include <hip/hip_bf16.h>

typedef _Float16 v8h  __attribute__((ext_vector_type(8)));
typedef _Float16 v16h __attribute__((ext_vector_type(16)));
typedef float    v8f  __attribute__((ext_vector_type(8)));

#define BSZ   4096
#define DIM   256
#define NCTR  8192            // 2 * BSZ contrast rows
#define NTILE (NCTR / 16)     // 512 column tiles of 16
#define NITER (NTILE / 8)     // 64 tiles per wave (8 waves/block)
#define INV_T (1.0f / 0.07f)
#define EPSV  1e-10f

// ---------------------------------------------------------------------------
// Kernel 1: convert features (bsz, 2, 256) f32 -> contrastH (8192, 256) f16.
// contrast row r<4096 = features[r, 0, :], row r>=4096 = features[r-4096, 1, :].
// ---------------------------------------------------------------------------
__global__ void convert_kernel(const float* __restrict__ feat,
                               _Float16* __restrict__ ctr) {
  int idx  = blockIdx.x * blockDim.x + threadIdx.x;   // 0 .. 8192*256-1
  int r    = idx >> 8;                                // row 0..8191
  int dcol = idx & 255;
  int b    = r & (BSZ - 1);
  int v    = r >> 12;                                 // 0 or 1
  ctr[idx] = (_Float16)feat[(b << 9) + (v << 8) + dcol];
}

// ---------------------------------------------------------------------------
// Kernel 2: fused GEMM + masked log-softmax statistics.
// One block = 16 anchor rows; 8 waves stride over 512 column tiles.
// B fragments are register double-buffered one tile ahead; epilogue metadata
// is prefetched one tile ahead (issued before the B loads, so it is always
// landed by consume time given in-order load completion).
// ---------------------------------------------------------------------------
__launch_bounds__(256)
__global__ void supcon_main(const _Float16* __restrict__ ctr,
                            const float* __restrict__ va,
                            const int* __restrict__ dia,
                            const int* __restrict__ thresPtr,
                            float* __restrict__ partials) {
  const int lane    = threadIdx.x & 31;
  const int wave    = threadIdx.x >> 5;
  const int laneMod = lane & 15;
  const int hi      = lane >> 4;               // 0: lanes 0-15, 1: lanes 16-31
  const int rowBase = blockIdx.x << 4;         // 16 rows per block
  const int rBase8  = rowBase + hi * 8;        // this lane's 8 C-matrix rows

  const float th  = (float)thresPtr[0];
  const float th2 = th * th;                   // euclid < th  <=>  dist2 < th2

  __shared__ float redE[8][16];
  __shared__ float redL[8][16];
  __shared__ float redM[8][16];
  __shared__ float rowC[16];

  // Per-lane row metadata for its 8 M rows (C layout: M = vgpr + 8*hi).
  int   diaR[8]; float va0R[8], va1R[8];
#pragma unroll
  for (int m = 0; m < 8; ++m) {
    int i   = rBase8 + m;
    diaR[m] = dia[i];
    va0R[m] = va[2 * i];
    va1R[m] = va[2 * i + 1];
  }

  // A fragments, kept in registers for all 8 K-steps (K = 256 = 8 * 32).
  // 16-bit A 16x32 layout: lane m<16 holds K {0..7, 16..23}; lanes 16-31
  // hold K {8..15, 24..31} (two contiguous 16-byte chunks per lane).
  v16h afrag[8];
  {
    const _Float16* arow = ctr + (size_t)(rowBase + laneMod) * DIM;
#pragma unroll
    for (int ks = 0; ks < 8; ++ks) {
      int kb  = ks * 32 + hi * 8;
      v8h lo  = *(const v8h*)(arow + kb);
      v8h hh  = *(const v8h*)(arow + kb + 16);
      afrag[ks] = __builtin_shufflevector(lo, hh,
          0, 1, 2, 3, 4, 5, 6, 7, 8, 9, 10, 11, 12, 13, 14, 15);
    }
  }

  float sExp[8] = {0.f, 0.f, 0.f, 0.f, 0.f, 0.f, 0.f, 0.f};
  float sML[8]  = {0.f, 0.f, 0.f, 0.f, 0.f, 0.f, 0.f, 0.f};
  float sM[8]   = {0.f, 0.f, 0.f, 0.f, 0.f, 0.f, 0.f, 0.f};

  // --- helpers -------------------------------------------------------------
  // Load the epilogue metadata for tile t (issued BEFORE that tile's B loads).
  auto loadMeta = [&](int t, int& diaC, float& vc0, float& vc1) {
    int jj = ((t << 4) + laneMod) & (BSZ - 1);
    diaC   = dia[jj];
    float2 v2 = *(const float2*)(va + 2 * jj);
    vc0 = v2.x; vc1 = v2.y;
  };

  // Load the 8 B fragments for tile t.
  // 16-bit B 32x16 layout: lanes 0-15 hold K {0..7, 16..23} of column N=lane,
  // lanes 16-31 hold K {8..15, 24..31} (two 16-byte chunks).
  auto loadB = [&](int t, v16h* buf) {
    const _Float16* brow = ctr + (size_t)((t << 4) + laneMod) * DIM;
#pragma unroll
    for (int ks = 0; ks < 8; ++ks) {
      int kb = ks * 32 + hi * 8;
      v8h lo = *(const v8h*)(brow + kb);
      v8h hh = *(const v8h*)(brow + kb + 16);
      buf[ks] = __builtin_shufflevector(lo, hh,
          0, 1, 2, 3, 4, 5, 6, 7, 8, 9, 10, 11, 12, 13, 14, 15);
    }
  };

  // WMMA chain + masked softmax epilogue for tile t.
  auto computeTile = [&](int t, const v16h* buf, int diaC, float vc0, float vc1) {
    v8f c = {0.f, 0.f, 0.f, 0.f, 0.f, 0.f, 0.f, 0.f};
#pragma unroll
    for (int ks = 0; ks < 8; ++ks) {
      c = __builtin_amdgcn_wmma_f32_16x16x32_f16(
          /*neg_a=*/false, afrag[ks], /*neg_b=*/false, buf[ks],
          /*c_mod=*/(short)0, c, /*reuse_a=*/false, /*reuse_b=*/false);
    }
    const int  j  = (t << 4) + laneMod;        // this lane's column (N)
    const int  jj = j & (BSZ - 1);
    const bool firstHalf = (t < (NTILE / 2));  // j < 4096
#pragma unroll
    for (int m = 0; m < 8; ++m) {
      int   i  = rBase8 + m;
      float ls = (c[m] - 1.0f) * INV_T;                  // logits - max (=1/T)
      float lm = (firstHalf && (j == i)) ? 0.0f : 1.0f;  // logits_mask
      float e  = __expf(ls) * lm;
      float d0 = va0R[m] - vc0;
      float d1 = va1R[m] - vc1;
      bool  mb = (i == jj) | (diaR[m] == diaC) |
                 (fmaf(d1, d1, d0 * d0) < th2);
      float mt = mb ? lm : 0.0f;                         // mask_t entry
      sExp[m] += e;
      sML[m]   = fmaf(mt, ls, sML[m]);
      sM[m]   += mt;
    }
  };

  // Prefetch B rows for tile t (row = 512 B; hint two tiles ahead).
  auto prefetchB = [&](int t) {
    const char* p = (const char*)(ctr + (size_t)((t << 4) + laneMod) * DIM);
    __builtin_prefetch(p, 0, 3);
    __builtin_prefetch(p + 256, 0, 3);
  };

  // --- software-pipelined main loop (unrolled x2: no dynamic buffer index) --
  v16h b0[8], b1[8];
  int   diaC0, diaC1;
  float vc00, vc01, vc10, vc11;

  loadMeta(wave, diaC0, vc00, vc01);
  loadB(wave, b0);

  for (int it = 0; it < NITER; it += 2) {
    const int t = wave + it * 8;

    // phase 0: consume b0 (tile t), fill b1 (tile t+8)   [it+1 < 64 always]
    loadMeta(t + 8, diaC1, vc10, vc11);
    loadB(t + 8, b1);
    if (it + 2 < NITER) prefetchB(t + 16);
    computeTile(t, b0, diaC0, vc00, vc01);

    // phase 1: consume b1 (tile t+8), fill b0 (tile t+16)
    if (it + 2 < NITER) {
      loadMeta(t + 16, diaC0, vc00, vc01);
      loadB(t + 16, b0);
      if (it + 3 < NITER) prefetchB(t + 24);
    }
    computeTile(t + 8, b1, diaC1, vc10, vc11);
  }

  // Reduce over the 16 lanes of each half-wave (the 16 N columns).
#pragma unroll
  for (int off = 1; off <= 8; off <<= 1) {
#pragma unroll
    for (int m = 0; m < 8; ++m) {
      sExp[m] += __shfl_xor(sExp[m], off, 32);
      sML[m]  += __shfl_xor(sML[m],  off, 32);
      sM[m]   += __shfl_xor(sM[m],   off, 32);
    }
  }

  // One lane per half-wave stages its 8 row sums to LDS.
  if (laneMod == 0) {
#pragma unroll
    for (int m = 0; m < 8; ++m) {
      redE[wave][hi * 8 + m] = sExp[m];
      redL[wave][hi * 8 + m] = sML[m];
      redM[wave][hi * 8 + m] = sM[m];
    }
  }
  __syncthreads();

  // Deterministic cross-wave reduction + per-row finalize.
  if (threadIdx.x < 16) {
    float e = 0.f, l = 0.f, mm = 0.f;
#pragma unroll
    for (int w = 0; w < 8; ++w) {
      e  += redE[w][threadIdx.x];
      l  += redL[w][threadIdx.x];
      mm += redM[w][threadIdx.x];
    }
    float mlpp = (l - __logf(e + EPSV) * mm) / (mm + EPSV);
    rowC[threadIdx.x] = -mlpp;
  }
  __syncthreads();

  if (threadIdx.x == 0) {
    float s = 0.f;
#pragma unroll
    for (int r = 0; r < 16; ++r) s += rowC[r];
    partials[blockIdx.x] = s;
  }
}

// ---------------------------------------------------------------------------
// Kernel 3: deterministic final sum of 256 block partials -> loss scalar.
// ---------------------------------------------------------------------------
__global__ void finish_kernel(const float* __restrict__ partials,
                              float* __restrict__ out) {
  if (threadIdx.x == 0 && blockIdx.x == 0) {
    float s = 0.f;
    for (int i = 0; i < BSZ / 16; ++i) s += partials[i];
    out[0] = s * (1.0f / (float)BSZ);
  }
}

extern "C" void kernel_launch(void* const* d_in, const int* in_sizes, int n_in,
                              void* d_out, int out_size, void* d_ws, size_t ws_size,
                              hipStream_t stream) {
  const float* feat  = (const float*)d_in[0];  // (4096, 2, 256) f32
  const float* va    = (const float*)d_in[1];  // (4096, 2) f32
  const int*   dia   = (const int*)d_in[2];    // (4096,) int
  const int*   thres = (const int*)d_in[3];    // scalar int

  _Float16* ctr      = (_Float16*)d_ws;        // 8192*256 f16 = 4 MB
  float*    partials = (float*)((char*)d_ws +
                        (size_t)NCTR * DIM * sizeof(_Float16));  // 256 f32

  convert_kernel<<<(NCTR * DIM) / 256, 256, 0, stream>>>(feat, ctr);
  supcon_main<<<BSZ / 16, 256, 0, stream>>>(ctr, va, dia, thres, partials);
  finish_kernel<<<1, 32, 0, stream>>>(partials, (float*)d_out);
}
